// MLAAttention_29944511988380
// MI455X (gfx1250) — matrix operand order repaired and checked by reading, not compiled, not run on hardware
//
#include <hip/hip_runtime.h>

#define D_MODEL 1024
#define N_HEADS 16
#define D_HEAD  64
#define LATENT  32
#define BATCH   2
#define SEQ     4096
#define ROWS    (BATCH * SEQ)   // 8192

typedef __attribute__((ext_vector_type(16))) __bf16 v16bf;
typedef __attribute__((ext_vector_type(8)))  __bf16 v8bf;
typedef __attribute__((ext_vector_type(8)))  float  v8f;

// ---------------------------------------------------------------- helpers ---

__device__ __forceinline__ __bf16 f32_to_bf16(float f) {
  union { float f; unsigned int u; } v; v.f = f;
  unsigned int r = v.u + 0x7FFFu + ((v.u >> 16) & 1u);  // round-to-nearest-even
  unsigned short h = (unsigned short)(r >> 16);
  __bf16 b; __builtin_memcpy(&b, &h, sizeof(h));
  return b;
}

__device__ __forceinline__ void zero16B(__bf16* d) {
  *reinterpret_cast<uint4*>(d) = make_uint4(0u, 0u, 0u, 0u);
}

// CDNA5 async global->LDS copy (16B per lane), tracked by ASYNCcnt.
// VDST = LDS byte address (low 32 bits of flat shared address), VADDR = 64-bit global addr.
__device__ __forceinline__ void async_load_b128(__bf16* lds_dst, const __bf16* gsrc) {
  asm volatile("global_load_async_to_lds_b128 %0, %1, off"
               :: "v"((unsigned)(unsigned long long)lds_dst),
                  "v"((unsigned long long)gsrc)
               : "memory");
}

__device__ __forceinline__ void wait_async0() {
#if __has_builtin(__builtin_amdgcn_s_wait_asynccnt)
  __builtin_amdgcn_s_wait_asynccnt(0);
#else
  asm volatile("s_wait_asynccnt 0x0" ::: "memory");
#endif
}

// A-fragment (16x32 bf16, MxK): lane = m (0..15), elems 0..7 -> k = 8*(lane>=16)+e,
// elems 8..15 -> k = 16 + 8*(lane>=16)+e   (ISA 7.12.2, 16-bit A 16x32)
__device__ __forceinline__ v16bf load_a_frag(const __bf16* base, int ld, int lane) {
  const int m  = lane & 15;
  const int k8 = (lane & 16) ? 8 : 0;
  const __bf16* p = base + m * ld + k8;
  v8bf lo = *reinterpret_cast<const v8bf*>(p);
  v8bf hi = *reinterpret_cast<const v8bf*>(p + 16);
  v16bf r;
#pragma unroll
  for (int i = 0; i < 8; ++i) { r[i] = lo[i]; r[8 + i] = hi[i]; }
  return r;
}

// B-fragment (32x16 bf16, KxN) from a TRANSPOSED tile (rows = n, cols = k):
// lane = n (0..15), elems 0..7 -> k = 8*(lane>=16)+e, elems 8..15 -> k = 16 + 8*(lane>=16)+e.
__device__ __forceinline__ v16bf load_b_frag(const __bf16* baseT, int ld, int lane) {
  const int n  = lane & 15;
  const int k0 = (lane & 16) ? 8 : 0;
  const __bf16* p = baseT + n * ld + k0;
  v8bf lo = *reinterpret_cast<const v8bf*>(p);
  v8bf hi = *reinterpret_cast<const v8bf*>(p + 16);
  v16bf r;
#pragma unroll
  for (int i = 0; i < 8; ++i) { r[i] = lo[i]; r[8 + i] = hi[i]; }
  return r;
}

__device__ __forceinline__ v8f wmma_bf16f32(v16bf a, v16bf b, v8f c) {
  return __builtin_amdgcn_wmma_f32_16x16x32_bf16(
      false, a, false, b, (short)0, c, false, false);
}

// ------------------------------------------------------------- conversions ---

__global__ void cvt_bf16_kernel(const float* __restrict__ src,
                                __bf16* __restrict__ dst, int n) {
  int i = blockIdx.x * blockDim.x + threadIdx.x;
  if (i < n) dst[i] = f32_to_bf16(src[i]);
}

// src [R][C] row-major  ->  dst [C][R] row-major (transpose + bf16)
__global__ void cvt_t_bf16_kernel(const float* __restrict__ src,
                                  __bf16* __restrict__ dst, int R, int C) {
  int i = blockIdx.x * blockDim.x + threadIdx.x;
  if (i < R * C) {
    int r = i / C, c = i - r * C;
    dst[(size_t)c * R + r] = f32_to_bf16(src[i]);
  }
}

// ------------------------------------------------------------------- GEMM ---
// C[M,N] = A[M,K] * B[K,N], B supplied transposed: Bt[N,K].
// 128 threads (4 waves). Block tile 64x64, wave tile 32x32 (2x2 WMMA).
// Double-buffered LDS with async global->LDS staging.

template <bool OUT_BF16>
__global__ void __launch_bounds__(128)
gemm_bf16_kernel(const __bf16* __restrict__ A, int lda,
                 const __bf16* __restrict__ Bt, int ldbt,
                 void* __restrict__ Cp, int ldc,
                 int M, int N, int K) {
  constexpr int LD   = 72;       // padded LDS leading dim (16B-aligned rows)
  constexpr int TILE = 64 * LD;
  __shared__ __align__(16) __bf16 As[2 * TILE];
  __shared__ __align__(16) __bf16 Bs[2 * TILE];

  const int tid  = threadIdx.x;
  const int lane = tid & 31;
  const int wave = tid >> 5;
  const int m0 = blockIdx.y * 64;
  const int n0 = blockIdx.x * 64;
  const int wm = (wave & 1) * 32;
  const int wn = (wave >> 1) * 32;

  const int srow = tid >> 1;       // staging: 2 threads per row
  const int scol = (tid & 1) * 32;
  const bool nv = (n0 + srow) < N;

  auto stage = [&](int kt, int buf) {
    const int kbase = kt * 64;
    const __bf16* gA = A  + (size_t)(m0 + srow) * lda  + kbase + scol;
    const __bf16* gB = Bt + (size_t)(n0 + srow) * ldbt + kbase + scol;
    __bf16* da = &As[buf * TILE + srow * LD + scol];
    __bf16* db = &Bs[buf * TILE + srow * LD + scol];
#pragma unroll
    for (int i = 0; i < 4; ++i) {
      const int k = kbase + scol + i * 8;
      if (k + 8 <= K) async_load_b128(da + i * 8, gA + i * 8); else zero16B(da + i * 8);
      if (nv && (k + 8 <= K)) async_load_b128(db + i * 8, gB + i * 8); else zero16B(db + i * 8);
    }
  };

  v8f acc00 = {}, acc01 = {}, acc10 = {}, acc11 = {};

  const int ktiles = (K + 63) / 64;
  stage(0, 0);
  wait_async0();
  __syncthreads();

  for (int kt = 0; kt < ktiles; ++kt) {
    const int cur = kt & 1;
    if (kt + 1 < ktiles) stage(kt + 1, cur ^ 1);   // prefetch next tile (async)

    const __bf16* Ab = &As[cur * TILE];
    const __bf16* Bb = &Bs[cur * TILE];
#pragma unroll
    for (int ks = 0; ks < 64; ks += 32) {
      v16bf a0 = load_a_frag(&Ab[(wm +  0) * LD + ks], LD, lane);
      v16bf a1 = load_a_frag(&Ab[(wm + 16) * LD + ks], LD, lane);
      v16bf b0 = load_b_frag(&Bb[(wn +  0) * LD + ks], LD, lane);
      v16bf b1 = load_b_frag(&Bb[(wn + 16) * LD + ks], LD, lane);
      acc00 = wmma_bf16f32(a0, b0, acc00);
      acc01 = wmma_bf16f32(a0, b1, acc01);
      acc10 = wmma_bf16f32(a1, b0, acc10);
      acc11 = wmma_bf16f32(a1, b1, acc11);
    }
    wait_async0();
    __syncthreads();
  }

  const int nloc = lane & 15;
  const int moff = (lane & 16) ? 8 : 0;
#pragma unroll
  for (int r = 0; r < 8; ++r) {
    const int mA = m0 + wm + r + moff;
    const int mB = mA + 16;
    const int nA = n0 + wn + nloc;
    const int nB = nA + 16;
    if (OUT_BF16) {
      __bf16* C = (__bf16*)Cp;
      if (nA < N) { C[(size_t)mA * ldc + nA] = f32_to_bf16(acc00[r]);
                    C[(size_t)mB * ldc + nA] = f32_to_bf16(acc10[r]); }
      if (nB < N) { C[(size_t)mA * ldc + nB] = f32_to_bf16(acc01[r]);
                    C[(size_t)mB * ldc + nB] = f32_to_bf16(acc11[r]); }
    } else {
      float* C = (float*)Cp;
      if (nA < N) { C[(size_t)mA * ldc + nA] = acc00[r];
                    C[(size_t)mB * ldc + nA] = acc10[r]; }
      if (nB < N) { C[(size_t)mA * ldc + nB] = acc01[r];
                    C[(size_t)mB * ldc + nB] = acc11[r]; }
    }
  }
}

// --------------------------------------------------------- flash attention ---
// grid = (SEQ/64, BATCH*N_HEADS), block = 128 (4 waves; wave w owns q rows w*16..+15)
// Q/K/V/O are [B, T, H*dh] bf16; head slice at column h*64.
// K and V^T tiles are double-buffered; K staged via async global->LDS.

__global__ void __launch_bounds__(128)
mla_flash_kernel(const __bf16* __restrict__ Qg, const __bf16* __restrict__ Kg,
                 const __bf16* __restrict__ Vg, __bf16* __restrict__ Og) {
  constexpr int LD   = 72;
  constexpr int TILE = 64 * LD;
  __shared__ __align__(16) __bf16 Qs [TILE];
  __shared__ __align__(16) __bf16 Ksh[2 * TILE];
  __shared__ __align__(16) __bf16 Vsh[2 * TILE];   // V transposed: [dh][kv]
  __shared__ __align__(16) __bf16 Ps [TILE];       // P tile (bf16) for 2nd WMMA

  const int tid  = threadIdx.x;
  const int lane = tid & 31;
  const int wave = tid >> 5;
  const int qt = blockIdx.x;
  const int bh = blockIdx.y;
  const int b  = bh / N_HEADS;
  const int h  = bh - b * N_HEADS;
  const int q0 = qt * 64;

  const size_t base = (size_t)b * SEQ * D_MODEL + (size_t)h * D_HEAD;
  const __bf16* Qp = Qg + base;
  const __bf16* Kp = Kg + base;
  const __bf16* Vp = Vg + base;
  __bf16*       Op = Og + base;

  const int srow = tid >> 1;        // staging helpers: 2 threads per row
  const int scol = (tid & 1) * 32;

  auto stageK = [&](int jt, int buf) {   // async: rows of K
    const __bf16* g = Kp + (size_t)(jt * 64 + srow) * D_MODEL + scol;
    __bf16* d = &Ksh[buf * TILE + srow * LD + scol];
#pragma unroll
    for (int i = 0; i < 4; ++i) async_load_b128(d + i * 8, g + i * 8);
  };
  auto stageVt = [&](int jt, int buf) {  // coalesced row read + transposed LDS scatter
    const __bf16* g = Vp + (size_t)(jt * 64 + srow) * D_MODEL + scol;
    __bf16* d = &Vsh[buf * TILE];
#pragma unroll
    for (int c = 0; c < 4; ++c) {
      v8bf t = *reinterpret_cast<const v8bf*>(g + c * 8);
#pragma unroll
      for (int j = 0; j < 8; ++j) d[(scol + c * 8 + j) * LD + srow] = t[j];
    }
  };

  // prologue: Q (async), K/Vt tile 0
  {
    const __bf16* g = Qp + (size_t)(q0 + srow) * D_MODEL + scol;
#pragma unroll
    for (int i = 0; i < 4; ++i) async_load_b128(&Qs[srow * LD + scol + i * 8], g + i * 8);
  }
  stageK(0, 0);
  stageVt(0, 0);
  wait_async0();
  __syncthreads();

  const int nloc = lane & 15;
  const int moff = (lane & 16) ? 8 : 0;
  const float scale = 0.125f;               // 1/sqrt(64)
  const float L2E   = 1.44269504088896f;

  float mrow[8], lrow[8];
#pragma unroll
  for (int r = 0; r < 8; ++r) { mrow[r] = -1.0e30f; lrow[r] = 0.0f; }
  v8f o0 = {}, o1 = {}, o2 = {}, o3 = {};

  for (int jt = 0; jt <= qt; ++jt) {        // causal: tiles at/below diagonal only
    const int kv0 = jt * 64;
    const int cur = jt & 1;
    if (jt < qt) { stageK(jt + 1, cur ^ 1); stageVt(jt + 1, cur ^ 1); }  // prefetch

    const __bf16* Kb = &Ksh[cur * TILE];
    const __bf16* Vb = &Vsh[cur * TILE];

    // S = Q * K^T  (4 n-subtiles of 16 kv columns each)
    v8f s0 = {}, s1 = {}, s2 = {}, s3 = {};
#pragma unroll
    for (int ks = 0; ks < 64; ks += 32) {
      v16bf aq  = load_a_frag(&Qs[(wave * 16) * LD + ks], LD, lane);
      v16bf k0f = load_b_frag(&Kb[ 0 * LD + ks], LD, lane);
      v16bf k1f = load_b_frag(&Kb[16 * LD + ks], LD, lane);
      v16bf k2f = load_b_frag(&Kb[32 * LD + ks], LD, lane);
      v16bf k3f = load_b_frag(&Kb[48 * LD + ks], LD, lane);
      s0 = wmma_bf16f32(aq, k0f, s0);
      s1 = wmma_bf16f32(aq, k1f, s1);
      s2 = wmma_bf16f32(aq, k2f, s2);
      s3 = wmma_bf16f32(aq, k3f, s3);
    }

    // online softmax; C-fragment: row m = r + moff, col n = nloc (+16*subtile)
    const bool diag = (jt == qt);
#pragma unroll
    for (int r = 0; r < 8; ++r) {
      const int qg = q0 + wave * 16 + r + moff;
      float v0 = s0[r] * scale, v1 = s1[r] * scale;
      float v2 = s2[r] * scale, v3 = s3[r] * scale;
      if (diag) {
        v0 = (kv0 +  0 + nloc > qg) ? -1.0e30f : v0;
        v1 = (kv0 + 16 + nloc > qg) ? -1.0e30f : v1;
        v2 = (kv0 + 32 + nloc > qg) ? -1.0e30f : v2;
        v3 = (kv0 + 48 + nloc > qg) ? -1.0e30f : v3;
      }
      float mx = fmaxf(fmaxf(v0, v1), fmaxf(v2, v3));
#pragma unroll
      for (int sh = 1; sh <= 8; sh <<= 1) mx = fmaxf(mx, __shfl_xor(mx, sh, 32));
      const float mnew  = fmaxf(mrow[r], mx);
      const float alpha = exp2f(L2E * (mrow[r] - mnew));
      const float p0 = exp2f(L2E * (v0 - mnew));
      const float p1 = exp2f(L2E * (v1 - mnew));
      const float p2 = exp2f(L2E * (v2 - mnew));
      const float p3 = exp2f(L2E * (v3 - mnew));
      float ps = p0 + p1 + p2 + p3;
#pragma unroll
      for (int sh = 1; sh <= 8; sh <<= 1) ps += __shfl_xor(ps, sh, 32);
      lrow[r] = lrow[r] * alpha + ps;
      mrow[r] = mnew;
      o0[r] *= alpha; o1[r] *= alpha; o2[r] *= alpha; o3[r] *= alpha;
      const int prow = wave * 16 + r + moff;
      Ps[prow * LD +  0 + nloc] = f32_to_bf16(p0);
      Ps[prow * LD + 16 + nloc] = f32_to_bf16(p1);
      Ps[prow * LD + 32 + nloc] = f32_to_bf16(p2);
      Ps[prow * LD + 48 + nloc] = f32_to_bf16(p3);
    }
    __syncthreads();   // Ps visible (wave-local ds write->read ordering)

    // O += P * V   (B-fragments from transposed V tile)
#pragma unroll
    for (int ks = 0; ks < 64; ks += 32) {
      v16bf ap  = load_a_frag(&Ps[(wave * 16) * LD + ks], LD, lane);
      v16bf b0f = load_b_frag(&Vb[ 0 * LD + ks], LD, lane);
      v16bf b1f = load_b_frag(&Vb[16 * LD + ks], LD, lane);
      v16bf b2f = load_b_frag(&Vb[32 * LD + ks], LD, lane);
      v16bf b3f = load_b_frag(&Vb[48 * LD + ks], LD, lane);
      o0 = wmma_bf16f32(ap, b0f, o0);
      o1 = wmma_bf16f32(ap, b1f, o1);
      o2 = wmma_bf16f32(ap, b2f, o2);
      o3 = wmma_bf16f32(ap, b3f, o3);
    }
    wait_async0();     // this wave's prefetch of K[nxt] landed
    __syncthreads();   // everyone done with cur buffers; nxt buffers published
  }

#pragma unroll
  for (int r = 0; r < 8; ++r) {
    const float inv = 1.0f / lrow[r];
    const int trow = q0 + wave * 16 + r + moff;
    __bf16* dst = Op + (size_t)trow * D_MODEL;
    dst[ 0 + nloc] = f32_to_bf16(o0[r] * inv);
    dst[16 + nloc] = f32_to_bf16(o1[r] * inv);
    dst[32 + nloc] = f32_to_bf16(o2[r] * inv);
    dst[48 + nloc] = f32_to_bf16(o3[r] * inv);
  }
}

// ------------------------------------------------------------------- host ---

extern "C" void kernel_launch(void* const* d_in, const int* in_sizes, int n_in,
                              void* d_out, int out_size, void* d_ws, size_t ws_size,
                              hipStream_t stream) {
  (void)in_sizes; (void)n_in; (void)out_size; (void)ws_size;

  const float* x   = (const float*)d_in[0];
  const float* Wkv = (const float*)d_in[1];
  const float* Wk  = (const float*)d_in[2];
  const float* Wv  = (const float*)d_in[3];
  const float* Wq  = (const float*)d_in[4];
  const float* Wo  = (const float*)d_in[5];
  float* out = (float*)d_out;

  char* p = (char*)d_ws;
  auto alloc = [&](size_t bytes) -> char* {
    char* r = p;
    p += (bytes + 255) & ~(size_t)255;
    return r;
  };
  __bf16* xb   = (__bf16*)alloc((size_t)ROWS * D_MODEL * 2);
  __bf16* wkvT = (__bf16*)alloc((size_t)LATENT * D_MODEL * 2);   // [32][1024]
  __bf16* wkT  = (__bf16*)alloc((size_t)D_MODEL * LATENT * 2);   // [1024][32]
  __bf16* wvT  = (__bf16*)alloc((size_t)D_MODEL * LATENT * 2);   // [1024][32]
  __bf16* wqT  = (__bf16*)alloc((size_t)D_MODEL * D_MODEL * 2);  // [1024][1024]
  __bf16* woT  = (__bf16*)alloc((size_t)D_MODEL * D_MODEL * 2);  // [1024][1024]
  __bf16* latb = (__bf16*)alloc((size_t)ROWS * LATENT * 2);
  __bf16* Qb   = (__bf16*)alloc((size_t)ROWS * D_MODEL * 2);
  __bf16* Kb   = (__bf16*)alloc((size_t)ROWS * D_MODEL * 2);
  __bf16* Vb   = (__bf16*)alloc((size_t)ROWS * D_MODEL * 2);
  __bf16* Ob   = (__bf16*)alloc((size_t)ROWS * D_MODEL * 2);

  cvt_bf16_kernel<<<(ROWS * D_MODEL) / 256, 256, 0, stream>>>(x, xb, ROWS * D_MODEL);
  cvt_t_bf16_kernel<<<(D_MODEL * LATENT + 255) / 256, 256, 0, stream>>>(Wkv, wkvT, D_MODEL, LATENT);
  cvt_t_bf16_kernel<<<(LATENT * D_MODEL + 255) / 256, 256, 0, stream>>>(Wk,  wkT,  LATENT, D_MODEL);
  cvt_t_bf16_kernel<<<(LATENT * D_MODEL + 255) / 256, 256, 0, stream>>>(Wv,  wvT,  LATENT, D_MODEL);
  cvt_t_bf16_kernel<<<(D_MODEL * D_MODEL + 255) / 256, 256, 0, stream>>>(Wq, wqT, D_MODEL, D_MODEL);
  cvt_t_bf16_kernel<<<(D_MODEL * D_MODEL + 255) / 256, 256, 0, stream>>>(Wo, woT, D_MODEL, D_MODEL);

  const dim3 blk(128);
  gemm_bf16_kernel<true><<<dim3(1, ROWS / 64), blk, 0, stream>>>(
      xb, D_MODEL, wkvT, D_MODEL, (void*)latb, LATENT, ROWS, LATENT, D_MODEL);
  gemm_bf16_kernel<true><<<dim3(D_MODEL / 64, ROWS / 64), blk, 0, stream>>>(
      latb, LATENT, wkT, LATENT, (void*)Kb, D_MODEL, ROWS, D_MODEL, LATENT);
  gemm_bf16_kernel<true><<<dim3(D_MODEL / 64, ROWS / 64), blk, 0, stream>>>(
      latb, LATENT, wvT, LATENT, (void*)Vb, D_MODEL, ROWS, D_MODEL, LATENT);
  gemm_bf16_kernel<true><<<dim3(D_MODEL / 64, ROWS / 64), blk, 0, stream>>>(
      xb, D_MODEL, wqT, D_MODEL, (void*)Qb, D_MODEL, ROWS, D_MODEL, D_MODEL);
  mla_flash_kernel<<<dim3(SEQ / 64, BATCH * N_HEADS), blk, 0, stream>>>(Qb, Kb, Vb, Ob);
  gemm_bf16_kernel<false><<<dim3(D_MODEL / 64, ROWS / 64), blk, 0, stream>>>(
      Ob, D_MODEL, woT, D_MODEL, (void*)out, D_MODEL, ROWS, D_MODEL, D_MODEL);
}
